// _BiMamba2BlockCfg_7919919694495
// MI455X (gfx1250) — matrix operand order repaired and checked by reading, not compiled, not run on hardware
//
#include <hip/hip_runtime.h>
#include <hip/hip_bf16.h>
#include <math.h>

// ---------------- problem constants ----------------
#define D_MODEL   512
#define D_STATE   64
#define PDIM      64
#define QCHUNK    64
#define D_INNER   1024
#define NH        16          // D_INNER / PDIM
#define CONV_CH   1152        // D_INNER + 2*D_STATE
#define D_IN_PROJ 2192        // 2*D_INNER + 2*D_STATE + NH
#define NPROJ_PAD 2304        // 18 * 128 (tile-padded rows of in_w)
#define B_SZ      4
#define L_SEQ     2048
#define NCHUNK    32          // L_SEQ / QCHUNK
#define NROWS     (B_SZ * L_SEQ)   // 8192
#define EPS_RMS   1e-5f

typedef __bf16 bf16;
typedef bf16  v16bf __attribute__((ext_vector_type(16)));
typedef float v8f   __attribute__((ext_vector_type(8)));

__device__ __forceinline__ float siluf(float v)     { return v / (1.0f + __expf(-v)); }
__device__ __forceinline__ float softplusf(float v) { return (v > 20.0f) ? v : log1pf(__expf(v)); }

__device__ __forceinline__ v8f zero8() {
  v8f z;
#pragma unroll
  for (int i = 0; i < 8; ++i) z[i] = 0.0f;
  return z;
}

// ---------------- CDNA5 async global->LDS copy (16B per lane) ----------------
__device__ __forceinline__ void async_g2l_b128(void* lds, const void* gaddr) {
  unsigned lds_off = (unsigned)(unsigned long long)lds;  // LDS aperture: addr[31:0]
  asm volatile("global_load_async_to_lds_b128 %0, %1, off"
               :: "v"(lds_off), "v"(gaddr)
               : "memory");
}
__device__ __forceinline__ void async_wait0() {
  asm volatile("s_wait_asynccnt 0x0" ::: "memory");
}

// ---------------- WMMA bf16 16x16x32 fragment helpers ----------------
// A-fragment: element (m,k) at src[m*rs + k*cs].  Contiguous k order: k = 16*hh + s (matches frag_b_t).
__device__ __forceinline__ v16bf frag_a(const bf16* s, int rs, int cs) {
  int lane = threadIdx.x & 31, hh = lane >> 4, m = lane & 15;
  v16bf a;
#pragma unroll
  for (int i = 0; i < 8; ++i) {
    int kb = 2 * i + 16 * hh;
    a[2 * i]     = s[m * rs + kb * cs];
    a[2 * i + 1] = s[m * rs + (kb + 1) * cs];
  }
  return a;
}
// B-fragment from [col][k]-oriented LDS tile: element (k,n) at src[n*rs + k].
// Per-lane data is 16 contiguous bf16 -> 2x ds_load_b128.
__device__ __forceinline__ v16bf frag_b_t(const bf16* s, int rs) {
  int lane = threadIdx.x & 31, hh = lane >> 4, n = lane & 15;
  const bf16* p = s + n * rs + 16 * hh;
  v16bf b;
#pragma unroll
  for (int i = 0; i < 16; ++i) b[i] = p[i];
  return b;
}
__device__ __forceinline__ v8f wmma_bf16(v16bf a, v16bf b, v8f c) {
  return __builtin_amdgcn_wmma_f32_16x16x32_bf16(false, a, false, b, (short)0, c,
                                                 false, false);
}
// C/D fragment store: row m = r + 8*(lane>=16), col n = lane&15
__device__ __forceinline__ void frag_store(float* d, int rs, const v8f& c) {
  int lane = threadIdx.x & 31, hh = lane >> 4, n = lane & 15;
#pragma unroll
  for (int r = 0; r < 8; ++r) d[(r + 8 * hh) * rs + n] = c[r];
}
__device__ __forceinline__ void frag_store_add(float* d, int rs, const v8f& c) {
  int lane = threadIdx.x & 31, hh = lane >> 4, n = lane & 15;
#pragma unroll
  for (int r = 0; r < 8; ++r) d[(r + 8 * hh) * rs + n] += c[r];
}
__device__ __forceinline__ void frag_store_add_scaled(float* d, int rs,
                                                      const float* scale, const v8f& c) {
  int lane = threadIdx.x & 31, hh = lane >> 4, n = lane & 15;
#pragma unroll
  for (int r = 0; r < 8; ++r) {
    int m = r + 8 * hh;
    d[m * rs + n] += c[r] * scale[m];
  }
}

// ---------------- Kernel 0: f32 -> bf16 conversion (with zero pad) ----------------
__global__ void k_cvt(const float* __restrict__ src, bf16* __restrict__ dst,
                      long long n_src, long long n_dst) {
  long long i = (long long)blockIdx.x * blockDim.x + threadIdx.x;
  if (i >= n_dst) return;
  dst[i] = (i < n_src) ? (bf16)src[i] : (bf16)0.0f;
}

// ---------------- Kernel 1: in-projection GEMM (both directions) ----------------
// proj[dir][row][0..2192) = x_dir[row] @ in_w^T   (bwd: x rows flipped per sequence)
__global__ __launch_bounds__(256)
void k_inproj(const bf16* __restrict__ xbf,
              const bf16* __restrict__ inwbf,   // [2][NPROJ_PAD][D_MODEL]
              float* __restrict__ proj) {
  const int dir = blockIdx.z;
  const int m0  = blockIdx.y * 128;
  const int n0  = blockIdx.x * 128;
  const bf16* W = inwbf + (size_t)dir * NPROJ_PAD * D_MODEL;
  __shared__ __align__(16) bf16 As[128][40];   // [m][k], 80B row stride
  __shared__ __align__(16) bf16 BsT[128][40];  // [n][k]
  const int tid = threadIdx.x, wave = tid >> 5;
  const int wm = wave >> 2, wn = wave & 3;

  v8f acc[4][2];
#pragma unroll
  for (int i = 0; i < 4; ++i)
#pragma unroll
    for (int j = 0; j < 2; ++j) acc[i][j] = zero8();

  for (int k0 = 0; k0 < D_MODEL; k0 += 32) {
    for (int i = tid; i < 512; i += 256) {       // A tile: 128 rows x 4 16B chunks
      int r = i >> 2, cc = i & 3;
      int row = m0 + r;
      int b = row >> 11, l = row & 2047;
      int srow = dir ? (b * L_SEQ + (L_SEQ - 1 - l)) : row;
      async_g2l_b128(&As[r][cc * 8], &xbf[(size_t)srow * D_MODEL + k0 + cc * 8]);
    }
    for (int i = tid; i < 512; i += 256) {       // B tile: 128 cols x 4 16B chunks
      int nn = i >> 2, cc = i & 3;
      async_g2l_b128(&BsT[nn][cc * 8], &W[(size_t)(n0 + nn) * D_MODEL + k0 + cc * 8]);
    }
    async_wait0();
    __syncthreads();
    v16bf af[4], bfr[2];
#pragma unroll
    for (int i = 0; i < 4; ++i) af[i] = frag_a(&As[wm * 64 + i * 16][0], 40, 1);
#pragma unroll
    for (int j = 0; j < 2; ++j) bfr[j] = frag_b_t(&BsT[wn * 32 + j * 16][0], 40);
#pragma unroll
    for (int i = 0; i < 4; ++i)
#pragma unroll
      for (int j = 0; j < 2; ++j) acc[i][j] = wmma_bf16(af[i], bfr[j], acc[i][j]);
    __syncthreads();
  }
#pragma unroll
  for (int i = 0; i < 4; ++i)
#pragma unroll
    for (int j = 0; j < 2; ++j) {
      int mrow = m0 + wm * 64 + i * 16;
      int ncol = n0 + wn * 32 + j * 16;
      if (ncol < D_IN_PROJ)
        frag_store(&proj[(size_t)dir * NROWS * D_IN_PROJ +
                         (size_t)mrow * D_IN_PROJ + ncol],
                   D_IN_PROJ, acc[i][j]);
    }
}

// ---------------- Kernel 2a: dt softplus and a_bar ----------------
__global__ void k_dt(const float* __restrict__ proj,
                     const float* dtb_f, const float* dtb_b,
                     const float* Al_f, const float* Al_b,
                     float* __restrict__ dtv, float* __restrict__ abar) {
  int i = blockIdx.x * blockDim.x + threadIdx.x;
  if (i >= 2 * NROWS * NH) return;
  int h = i % NH, row = (i / NH) % NROWS, dir = i / (NH * NROWS);
  const float* dtb = dir ? dtb_b : dtb_f;
  const float* Al  = dir ? Al_b : Al_f;
  float raw = proj[(size_t)dir * NROWS * D_IN_PROJ + (size_t)row * D_IN_PROJ +
                   (D_INNER + CONV_CH) + h];
  float d = softplusf(raw + dtb[h]);
  dtv[i]  = d;
  abar[i] = -__expf(Al[h]) * d;
}

// ---------------- Kernel 2b: causal depthwise conv + SiLU, split outputs ----------------
__global__ void k_conv(const float* __restrict__ proj,
                       const float* cw_f, const float* cw_b,
                       const float* cb_f, const float* cb_b,
                       const float* Dp_f, const float* Dp_b,
                       const float* __restrict__ dtv,
                       bf16* __restrict__ xdtbf, bf16* __restrict__ bcbf,
                       float* __restrict__ y) {
  long long i = (long long)blockIdx.x * blockDim.x + threadIdx.x;
  if (i >= (long long)2 * NROWS * CONV_CH) return;
  int c   = (int)(i % CONV_CH);
  int row = (int)((i / CONV_CH) % NROWS);
  int dir = (int)(i / ((long long)CONV_CH * NROWS));
  const float* cw = dir ? cw_b : cw_f;
  const float* cb = dir ? cb_b : cb_f;
  const float* Dp = dir ? Dp_b : Dp_f;
  int l = row & 2047;
  const float* prow = proj + (size_t)dir * NROWS * D_IN_PROJ +
                      (size_t)row * D_IN_PROJ + D_INNER + c;
  float acc = cb[c];
#pragma unroll
  for (int j = 0; j < 4; ++j) {
    int dl = l - 3 + j;
    if (dl >= 0) acc += prow[(long long)(j - 3) * D_IN_PROJ] * cw[c * 4 + j];
  }
  float v = siluf(acc);
  size_t base = (size_t)dir * NROWS + row;
  if (c < D_INNER) {
    int h = c >> 6;
    float d = dtv[((size_t)dir * NROWS + row) * NH + h];
    xdtbf[base * D_INNER + c] = (bf16)(v * d);
    y[base * D_INNER + c]     = Dp[h] * v;     // D * x  term (y init)
  } else {
    bcbf[base * 128 + (c - D_INNER)] = (bf16)v; // [0..63]=B, [64..127]=C
  }
}

// ---------------- Kernel 3: per-chunk cumulative decay ----------------
__global__ void k_cumsum(const float* __restrict__ abar,
                         float* __restrict__ acs, float* __restrict__ cdec) {
  int i = blockIdx.x * blockDim.x + threadIdx.x; // (dir,b,c,h)
  if (i >= 2 * B_SZ * NCHUNK * NH) return;
  int h = i % NH, c = (i / NH) % NCHUNK, b = (i / (NH * NCHUNK)) % B_SZ;
  float s = 0.0f;
  int dir = i / (NH * NCHUNK * B_SZ);
  for (int q = 0; q < QCHUNK; ++q) {
    int row = b * L_SEQ + c * QCHUNK + q;
    s += abar[((size_t)dir * NROWS + row) * NH + h];
    acs[(size_t)i * QCHUNK + q] = s;
  }
  cdec[i] = s;
}

// ---------------- Kernel 4: per (dir,b,c,h) diag block + chunk states (WMMA) ----------------
__global__ __launch_bounds__(256)
void k_chunk(const bf16* __restrict__ bcbf, const bf16* __restrict__ xdtbf,
             const float* __restrict__ acs,
             float* __restrict__ y, float* __restrict__ states) {
  int dir = blockIdx.z, b = blockIdx.y;
  int c = blockIdx.x >> 4, h = blockIdx.x & 15;
  __shared__ __align__(16) bf16 Cs[64][72];   // C tile; later reused as BdT[n][s]
  __shared__ __align__(16) bf16 Bs[64][72];   // B tile [s][n]
  __shared__ __align__(16) bf16 Xs[64][72];   // xdt tile [s][p]
  __shared__ __align__(16) bf16 XsT[64][72];  // [p][s]
  __shared__ __align__(16) bf16 Ms[64][72];   // masked-decay scores [l][s]
  __shared__ float sc[64][64];
  __shared__ float a_s[64];
  int tid = threadIdx.x, wave = tid >> 5;
  size_t rbase = (size_t)dir * NROWS + b * L_SEQ + c * QCHUNK;

  // stage B, C, X tiles via async copies (each row = 8 x 16B chunks)
  for (int i = tid; i < 512; i += 256) {
    int s = i >> 3, cc = i & 7;
    async_g2l_b128(&Bs[s][cc * 8], &bcbf[(rbase + s) * 128 + cc * 8]);
    async_g2l_b128(&Cs[s][cc * 8], &bcbf[(rbase + s) * 128 + 64 + cc * 8]);
    async_g2l_b128(&Xs[s][cc * 8], &xdtbf[(rbase + s) * D_INNER + h * 64 + cc * 8]);
  }
  int acsbase = ((((dir * B_SZ + b) * NCHUNK + c) * NH) + h) * QCHUNK;
  if (tid < 64) a_s[tid] = acs[acsbase + tid];
  async_wait0();
  __syncthreads();

  // build XsT[p][s]
  for (int i = tid; i < 64 * 64; i += 256) {
    int s = i >> 6, p = i & 63;
    XsT[p][s] = Xs[s][p];
  }
  __syncthreads();

  // scores = C @ B^T  (l x s'), K = n : B operand is Bs[s'][n] -> [col][k] layout
  for (int t = 0; t < 2; ++t) {
    int ti = wave * 2 + t, tm = ti >> 2, tn = ti & 3;
    v8f acc = zero8();
    for (int k0 = 0; k0 < 64; k0 += 32) {
      v16bf a   = frag_a(&Cs[tm * 16][k0], 72, 1);
      v16bf bfr = frag_b_t(&Bs[tn * 16][k0], 72);
      acc = wmma_bf16(a, bfr, acc);
    }
    frag_store(&sc[tm * 16][tn * 16], 64, acc);
  }
  __syncthreads();

  // masked decay M[l][s] ; decay-scaled B transposed BdT[n][s] (into Cs buffer)
  float dlast = a_s[63];
  for (int i = tid; i < 64 * 64; i += 256) {
    int l = i >> 6, s = i & 63;
    float v = (s <= l) ? sc[l][s] * __expf(a_s[l] - a_s[s]) : 0.0f;
    Ms[l][s] = (bf16)v;
    // i also enumerates (n = l, s): BdT[n][s] = exp(acs[63]-acs[s]) * B[s][n]
    Cs[l][s] = (bf16)(__expf(dlast - a_s[s]) * (float)Bs[s][l]);
  }
  __syncthreads();

  // y_diag = M @ X : B operand from XsT[p][s]  -> accumulate into y
  for (int t = 0; t < 2; ++t) {
    int ti = wave * 2 + t, tm = ti >> 2, tn = ti & 3;
    v8f acc = zero8();
    for (int k0 = 0; k0 < 64; k0 += 32) {
      v16bf a   = frag_a(&Ms[tm * 16][k0], 72, 1);
      v16bf bfr = frag_b_t(&XsT[tn * 16][k0], 72);
      acc = wmma_bf16(a, bfr, acc);
    }
    frag_store_add(&y[(rbase + tm * 16) * D_INNER + h * 64 + tn * 16],
                   D_INNER, acc);
  }

  // states S[p][n] = sum_s X[s][p] * Bd[s][n] : A from XsT, B from BdT (=Cs)
  size_t stbase = ((((size_t)dir * B_SZ + b) * NCHUNK + c) * NH + h) * 4096;
  for (int t = 0; t < 2; ++t) {
    int ti = wave * 2 + t, tm = ti >> 2, tn = ti & 3;
    v8f acc = zero8();
    for (int k0 = 0; k0 < 64; k0 += 32) {
      v16bf a   = frag_a(&XsT[tm * 16][k0], 72, 1);
      v16bf bfr = frag_b_t(&Cs[tn * 16][k0], 72);
      acc = wmma_bf16(a, bfr, acc);
    }
    frag_store(&states[stbase + (size_t)(tm * 16) * 64 + tn * 16], 64, acc);
  }
}

// ---------------- Kernel 5: sequential inter-chunk scan (in place: states -> prev) ---------
__global__ __launch_bounds__(256)
void k_scan(const float* __restrict__ cdec, float* __restrict__ states) {
  int h = blockIdx.x, b = blockIdx.y, dir = blockIdx.z;
  int tid = threadIdx.x;
  float carry[16];
#pragma unroll
  for (int e = 0; e < 16; ++e) carry[e] = 0.0f;
  for (int c = 0; c < NCHUNK; ++c) {
    float dec = cdec[((dir * B_SZ + b) * NCHUNK + c) * NH + h];
    size_t base = ((((size_t)dir * B_SZ + b) * NCHUNK + c) * NH + h) * 4096;
#pragma unroll
    for (int e = 0; e < 16; ++e) {
      int idx = e * 256 + tid;
      float s = states[base + idx];
      float prev = carry[e];
      carry[e] = prev * dec + s;
      states[base + idx] = prev;  // state entering this chunk
    }
  }
}

// ---------------- Kernel 6: y_off = exp(acs) * (C @ prev^T), accumulate (WMMA) ----------
__global__ __launch_bounds__(256)
void k_yoff(const bf16* __restrict__ bcbf, const float* __restrict__ states,
            const float* __restrict__ acs, float* __restrict__ y) {
  int dir = blockIdx.z, b = blockIdx.y;
  int c = blockIdx.x >> 4, h = blockIdx.x & 15;
  __shared__ __align__(16) bf16 Cs[64][72];
  __shared__ __align__(16) bf16 Ps[64][72];   // prev[p][n] -> [col=p][k=n]
  __shared__ float es[64];
  int tid = threadIdx.x, wave = tid >> 5;
  size_t rbase  = (size_t)dir * NROWS + b * L_SEQ + c * QCHUNK;
  size_t stbase = ((((size_t)dir * B_SZ + b) * NCHUNK + c) * NH + h) * 4096;
  for (int i = tid; i < 512; i += 256) {
    int r = i >> 3, cc = i & 7;
    async_g2l_b128(&Cs[r][cc * 8], &bcbf[(rbase + r) * 128 + 64 + cc * 8]);
  }
  for (int i = tid; i < 64 * 64; i += 256) {
    int r = i >> 6, n = i & 63;
    Ps[r][n] = (bf16)states[stbase + (size_t)r * 64 + n];
  }
  int acsbase = ((((dir * B_SZ + b) * NCHUNK + c) * NH) + h) * QCHUNK;
  if (tid < 64) es[tid] = __expf(acs[acsbase + tid]);
  async_wait0();
  __syncthreads();
  for (int t = 0; t < 2; ++t) {
    int ti = wave * 2 + t, tm = ti >> 2, tn = ti & 3;
    v8f acc = zero8();
    for (int k0 = 0; k0 < 64; k0 += 32) {
      v16bf a   = frag_a(&Cs[tm * 16][k0], 72, 1);
      v16bf bfr = frag_b_t(&Ps[tn * 16][k0], 72);
      acc = wmma_bf16(a, bfr, acc);
    }
    frag_store_add_scaled(&y[(rbase + tm * 16) * D_INNER + h * 64 + tn * 16],
                          D_INNER, &es[tm * 16], acc);
  }
}

// ---------------- Kernel 7: SiLU gate + RMSNorm (per row), emit bf16 ----------------
__global__ __launch_bounds__(256)
void k_gatenorm(const float* __restrict__ proj,
                const float* nw_f, const float* nw_b,
                const float* __restrict__ y, bf16* __restrict__ ybf) {
  int row = blockIdx.x, dir = blockIdx.y;
  const float* nw = dir ? nw_b : nw_f;
  __shared__ float red[256];
  size_t ybase = ((size_t)dir * NROWS + row) * D_INNER;
  size_t pbase = (size_t)dir * NROWS * D_IN_PROJ + (size_t)row * D_IN_PROJ;
  float vals[4];
  float ss = 0.0f;
#pragma unroll
  for (int e = 0; e < 4; ++e) {
    int ch = e * 256 + threadIdx.x;
    float g = siluf(proj[pbase + ch]);          // z gate
    float v = y[ybase + ch] * g;
    vals[e] = v;
    ss += v * v;
  }
  red[threadIdx.x] = ss;
  __syncthreads();
  for (int s = 128; s > 0; s >>= 1) {
    if (threadIdx.x < s) red[threadIdx.x] += red[threadIdx.x + s];
    __syncthreads();
  }
  float inv = rsqrtf(red[0] / (float)D_INNER + EPS_RMS);
#pragma unroll
  for (int e = 0; e < 4; ++e) {
    int ch = e * 256 + threadIdx.x;
    ybf[ybase + ch] = (bf16)(vals[e] * inv * nw[ch]);
  }
}

// ---------------- Kernel 8: out-projection GEMM, fwd + flipped bwd ----------------
__global__ __launch_bounds__(256)
void k_outproj(const bf16* __restrict__ ybf,
               const bf16* __restrict__ owbf,   // [2][D_MODEL][D_INNER]
               float* __restrict__ out) {
  int m0 = blockIdx.y * 128, n0 = blockIdx.x * 128;
  __shared__ __align__(16) bf16 As[128][40];
  __shared__ __align__(16) bf16 BsT[128][40];
  int tid = threadIdx.x, wave = tid >> 5, wm = wave >> 2, wn = wave & 3;
  v8f acc[4][2];
#pragma unroll
  for (int i = 0; i < 4; ++i)
#pragma unroll
    for (int j = 0; j < 2; ++j) acc[i][j] = zero8();

  for (int dir = 0; dir < 2; ++dir) {
    const bf16* W = owbf + (size_t)dir * D_MODEL * D_INNER;
    for (int k0 = 0; k0 < D_INNER; k0 += 32) {
      for (int i = tid; i < 512; i += 256) {
        int r = i >> 2, cc = i & 3;
        int orow = m0 + r;
        int b = orow >> 11, l = orow & 2047;
        int srow = dir ? (b * L_SEQ + (L_SEQ - 1 - l)) : orow;  // un-flip bwd
        async_g2l_b128(&As[r][cc * 8],
                       &ybf[((size_t)dir * NROWS + srow) * D_INNER + k0 + cc * 8]);
      }
      for (int i = tid; i < 512; i += 256) {
        int nn = i >> 2, cc = i & 3;
        async_g2l_b128(&BsT[nn][cc * 8], &W[(size_t)(n0 + nn) * D_INNER + k0 + cc * 8]);
      }
      async_wait0();
      __syncthreads();
      v16bf af[4], bfr[2];
#pragma unroll
      for (int i = 0; i < 4; ++i) af[i] = frag_a(&As[wm * 64 + i * 16][0], 40, 1);
#pragma unroll
      for (int j = 0; j < 2; ++j) bfr[j] = frag_b_t(&BsT[wn * 32 + j * 16][0], 40);
#pragma unroll
      for (int i = 0; i < 4; ++i)
#pragma unroll
        for (int j = 0; j < 2; ++j) acc[i][j] = wmma_bf16(af[i], bfr[j], acc[i][j]);
      __syncthreads();
    }
  }
#pragma unroll
  for (int i = 0; i < 4; ++i)
#pragma unroll
    for (int j = 0; j < 2; ++j)
      frag_store(&out[(size_t)(m0 + wm * 64 + i * 16) * D_MODEL +
                      n0 + wn * 32 + j * 16],
                 D_MODEL, acc[i][j]);
}

// ---------------- host launch ----------------
extern "C" void kernel_launch(void* const* d_in, const int* in_sizes, int n_in,
                              void* d_out, int out_size, void* d_ws, size_t ws_size,
                              hipStream_t stream) {
  (void)in_sizes; (void)n_in; (void)out_size; (void)ws_size;
  const float* x     = (const float*)d_in[0];
  const float* inw_f = (const float*)d_in[1];
  const float* cw_f  = (const float*)d_in[2];
  const float* cb_f  = (const float*)d_in[3];
  const float* dtb_f = (const float*)d_in[4];
  const float* Al_f  = (const float*)d_in[5];
  const float* Dp_f  = (const float*)d_in[6];
  const float* nw_f  = (const float*)d_in[7];
  const float* ow_f  = (const float*)d_in[8];
  const float* inw_b = (const float*)d_in[9];
  const float* cw_b  = (const float*)d_in[10];
  const float* cb_b  = (const float*)d_in[11];
  const float* dtb_b = (const float*)d_in[12];
  const float* Al_b  = (const float*)d_in[13];
  const float* Dp_b  = (const float*)d_in[14];
  const float* nw_b  = (const float*)d_in[15];
  const float* ow_b  = (const float*)d_in[16];
  float* out = (float*)d_out;

  // ---- workspace layout (~370 MB) ----
  char* p = (char*)d_ws;
  auto alloc = [&](size_t bytes) {
    void* q = (void*)p;
    p += (bytes + 255) & ~(size_t)255;
    return q;
  };
  float* proj   = (float*)alloc((size_t)2 * NROWS * D_IN_PROJ * 4);
  float* dtv    = (float*)alloc((size_t)2 * NROWS * NH * 4);
  float* abar   = (float*)alloc((size_t)2 * NROWS * NH * 4);
  float* acs    = (float*)alloc((size_t)2 * B_SZ * NCHUNK * NH * QCHUNK * 4);
  float* cdec   = (float*)alloc((size_t)2 * B_SZ * NCHUNK * NH * 4);
  float* states = (float*)alloc((size_t)2 * B_SZ * NCHUNK * NH * 4096 * 4);
  float* ybuf   = (float*)alloc((size_t)2 * NROWS * D_INNER * 4);
  bf16* xbf     = (bf16*)alloc((size_t)NROWS * D_MODEL * 2);
  bf16* inwbf   = (bf16*)alloc((size_t)2 * NPROJ_PAD * D_MODEL * 2);
  bf16* owbf    = (bf16*)alloc((size_t)2 * D_MODEL * D_INNER * 2);
  bf16* xdtbf   = (bf16*)alloc((size_t)2 * NROWS * D_INNER * 2);
  bf16* bcbf    = (bf16*)alloc((size_t)2 * NROWS * 128 * 2);
  bf16* ybf     = (bf16*)alloc((size_t)2 * NROWS * D_INNER * 2);

  // ---- bf16 conversions (weights padded to tile grid) ----
  const long long n_x   = (long long)NROWS * D_MODEL;
  const long long n_inw = (long long)D_IN_PROJ * D_MODEL;
  const long long n_inp = (long long)NPROJ_PAD * D_MODEL;
  const long long n_ow  = (long long)D_MODEL * D_INNER;
  k_cvt<<<dim3((unsigned)((n_x + 255) / 256)), 256, 0, stream>>>(x, xbf, n_x, n_x);
  k_cvt<<<dim3((unsigned)((n_inp + 255) / 256)), 256, 0, stream>>>(inw_f, inwbf, n_inw, n_inp);
  k_cvt<<<dim3((unsigned)((n_inp + 255) / 256)), 256, 0, stream>>>(inw_b, inwbf + n_inp, n_inw, n_inp);
  k_cvt<<<dim3((unsigned)((n_ow + 255) / 256)), 256, 0, stream>>>(ow_f, owbf, n_ow, n_ow);
  k_cvt<<<dim3((unsigned)((n_ow + 255) / 256)), 256, 0, stream>>>(ow_b, owbf + n_ow, n_ow, n_ow);

  // ---- pipeline ----
  k_inproj<<<dim3(NPROJ_PAD / 128, NROWS / 128, 2), 256, 0, stream>>>(xbf, inwbf, proj);
  k_dt<<<dim3((2 * NROWS * NH + 255) / 256), 256, 0, stream>>>(
      proj, dtb_f, dtb_b, Al_f, Al_b, dtv, abar);
  k_conv<<<dim3((2 * NROWS * CONV_CH + 255) / 256), 256, 0, stream>>>(
      proj, cw_f, cw_b, cb_f, cb_b, Dp_f, Dp_b, dtv, xdtbf, bcbf, ybuf);
  k_cumsum<<<dim3((2 * B_SZ * NCHUNK * NH + 255) / 256), 256, 0, stream>>>(
      abar, acs, cdec);
  k_chunk<<<dim3(NCHUNK * NH, B_SZ, 2), 256, 0, stream>>>(
      bcbf, xdtbf, acs, ybuf, states);
  k_scan<<<dim3(NH, B_SZ, 2), 256, 0, stream>>>(cdec, states);
  k_yoff<<<dim3(NCHUNK * NH, B_SZ, 2), 256, 0, stream>>>(
      bcbf, states, acs, ybuf);
  k_gatenorm<<<dim3(NROWS, 2), 256, 0, stream>>>(proj, nw_f, nw_b, ybuf, ybf);
  k_outproj<<<dim3(D_MODEL / 128, NROWS / 128), 256, 0, stream>>>(ybf, owbf, out);
}
